// Attention_38104949850738
// MI455X (gfx1250) — matrix operand order repaired and checked by reading, not compiled, not run on hardware
//
#include <hip/hip_runtime.h>

#define B_   256
#define L_   196
#define ENC_ 2048
#define DEC_ 512
#define ATT_ 512
#define ROWS_PER_WG 32
#define NROWS (B_ * L_)                 // 50176
#define NTILES (NROWS / ROWS_PER_WG)    // 1568 (exact)
#define APAD  8                         // LDS row pad (bf16 elems): stride%64banks==4
#define ASTRIDE (ENC_ + APAD)           // 2056 bf16 per LDS row

typedef __attribute__((ext_vector_type(16))) __bf16 v16bf;
typedef __attribute__((ext_vector_type(8)))  __bf16 v8bf;
typedef __attribute__((ext_vector_type(8)))  float  v8f;

__device__ __forceinline__ unsigned short f2bf(float f) {
  union { float f; unsigned u; } c; c.f = f;
  unsigned u = c.u;
  u += 0x7FFFu + ((u >> 16) & 1u);      // round-to-nearest-even
  return (unsigned short)(u >> 16);
}

// ---------------------------------------------------------------------------
// Kernel 1: W_enc (ENC x ATT fp32, K-major) -> wbT (ATT x ENC bf16, N-major)
// ---------------------------------------------------------------------------
__global__ __launch_bounds__(256)
void k_wenc_transpose(const float* __restrict__ W, unsigned short* __restrict__ out) {
  __shared__ float tile[32][33];
  const int n0 = blockIdx.x * 32;
  const int k0 = blockIdx.y * 32;
  const int tx = threadIdx.x & 31;
  const int ty = threadIdx.x >> 5;      // 0..7
#pragma unroll
  for (int i = 0; i < 32; i += 8)
    tile[ty + i][tx] = W[(size_t)(k0 + ty + i) * ATT_ + (n0 + tx)];
  __syncthreads();
#pragma unroll
  for (int i = 0; i < 32; i += 8)
    out[(size_t)(n0 + ty + i) * ENC_ + (k0 + tx)] = f2bf(tile[tx][ty + i]);
}

// ---------------------------------------------------------------------------
// Kernel 2: d[b,n] = dec[b,:] @ W_dec[:,n] + b_dec[n]   (256 x 512, tiny)
// ---------------------------------------------------------------------------
__global__ __launch_bounds__(256)
void k_dec_proj(const float* __restrict__ dec, const float* __restrict__ Wd,
                const float* __restrict__ bd, float* __restrict__ dOut) {
  __shared__ float sdec[DEC_];
  const int b = blockIdx.x, t = threadIdx.x;
  sdec[t]       = dec[b * DEC_ + t];
  sdec[t + 256] = dec[b * DEC_ + t + 256];
  __syncthreads();
#pragma unroll
  for (int h = 0; h < 2; ++h) {
    const int n = t + h * 256;
    float acc = bd[n];
    for (int k = 0; k < DEC_; ++k)
      acc = fmaf(sdec[k], Wd[(size_t)k * ATT_ + n], acc);
    dOut[b * ATT_ + n] = acc;
  }
}

// ---------------------------------------------------------------------------
// Kernel 3 (main): per-WG 32 rows x 512 cols of e = enc@W_enc via bf16 WMMA.
// A staged in LDS (bf16, natural K order per 32-block -> one contiguous 32B read
// per lane; padded row stride -> conflict-free banks). B fragments ping-pong
// double-buffered one K-step ahead of the WMMAs (no register copies).
// Fused epilogue: +b_enc +d[b] -> ReLU -> dot(W_full) -> raw att logits.
// ---------------------------------------------------------------------------
extern __shared__ unsigned short sA[];  // ROWS_PER_WG x ASTRIDE bf16 = 128.5 KB

__global__ __launch_bounds__(256, 2)
void k_enc_att(const float* __restrict__ enc,
               const unsigned short* __restrict__ wbT,
               const float* __restrict__ b_enc,
               const float* __restrict__ dmat,
               const float* __restrict__ W_full,
               const float* __restrict__ b_full,
               float* __restrict__ attRaw)
{
  __shared__ float sAtt[ROWS_PER_WG];
  const int tid  = threadIdx.x;
  const int lane = tid & 31;
  const int wv   = tid >> 5;            // 8 waves, 64 N-columns each
  const int row0 = blockIdx.x * ROWS_PER_WG;

  // ---- stage 32x2048 fp32 slab of enc into LDS as bf16, keeping each
  //      32-K block in natural order so lane halves read contiguously ----
  {
    const float4* src = reinterpret_cast<const float4*>(enc + (size_t)row0 * ENC_);
#pragma unroll 4
    for (int i = 0; i < (ROWS_PER_WG * ENC_ / 4) / 256; ++i) {
      const int f4  = i * 256 + tid;
      const float4 x = src[f4];
      const int e0  = f4 * 4;
      const int row = e0 >> 11;                 // /ENC_
      const int kk  = e0 & (ENC_ - 1);
      const int w32 = kk & 31;
      const int grp = w32 >> 3;
      const int dg  = grp;                                       // 0,1,2,3
      const int dpos = (kk & ~31) + dg * 8 + (w32 & 7);
      uint2 p;
      p.x = (unsigned)f2bf(x.x) | ((unsigned)f2bf(x.y) << 16);
      p.y = (unsigned)f2bf(x.z) | ((unsigned)f2bf(x.w) << 16);
      *reinterpret_cast<uint2*>(&sA[(size_t)row * ASTRIDE + dpos]) = p;
    }
  }
  if (tid < ROWS_PER_WG) sAtt[tid] = 0.0f;
  __syncthreads();

  const int nbase = wv * 64;
  const int mrow  = lane & 15;
  const int kh    = (lane >> 4) * 16;   // half-wave K selector (A and B)

  const unsigned short* bRow[4];
#pragma unroll
  for (int t = 0; t < 4; ++t)
    bRow[t] = wbT + (size_t)(nbase + t * 16 + mrow) * ENC_ + kh;

  const unsigned short* a0Base = &sA[(size_t)mrow * ASTRIDE + kh];
  const unsigned short* a1Base = &sA[(size_t)(16 + mrow) * ASTRIDE + kh];

  v8f acc0[4] = {};                     // rows [row0, row0+16)
  v8f acc1[4] = {};                     // rows [row0+16, row0+32)

  // ping-pong B fragment buffers; prologue fills set A with k = 0
  v16bf bA[4], bB[4];
#pragma unroll
  for (int t = 0; t < 4; ++t)
    bA[t] = *reinterpret_cast<const v16bf*>(bRow[t]);

  for (int k0 = 0; k0 < ENC_; k0 += 64) {
    // issue B loads for k0+32 (always in range: k0 <= ENC_-64)
#pragma unroll
    for (int t = 0; t < 4; ++t) {
      const char* p = reinterpret_cast<const char*>(bRow[t] + k0 + 32);
      __builtin_prefetch(p + 64, 0, 3);
      bB[t] = *reinterpret_cast<const v16bf*>(p);
    }
    {
      const v16bf a0 = *reinterpret_cast<const v16bf*>(a0Base + k0);
      const v16bf a1 = *reinterpret_cast<const v16bf*>(a1Base + k0);
#pragma unroll
      for (int t = 0; t < 4; ++t) {
        acc0[t] = __builtin_amdgcn_wmma_f32_16x16x32_bf16(
            false, a0, false, bA[t], (short)0, acc0[t], false, false);
        acc1[t] = __builtin_amdgcn_wmma_f32_16x16x32_bf16(
            false, a1, false, bA[t], (short)0, acc1[t], false, false);
      }
    }
    // issue B loads for k0+64 into the now-free set A
    const int kn = (k0 + 64 < ENC_) ? (k0 + 64) : k0;   // last iter: benign reload
#pragma unroll
    for (int t = 0; t < 4; ++t) {
      const char* p = reinterpret_cast<const char*>(bRow[t] + kn);
      __builtin_prefetch(p + 64, 0, 3);
      bA[t] = *reinterpret_cast<const v16bf*>(p);
    }
    {
      const v16bf a0 = *reinterpret_cast<const v16bf*>(a0Base + k0 + 32);
      const v16bf a1 = *reinterpret_cast<const v16bf*>(a1Base + k0 + 32);
#pragma unroll
      for (int t = 0; t < 4; ++t) {
        acc0[t] = __builtin_amdgcn_wmma_f32_16x16x32_bf16(
            false, a0, false, bB[t], (short)0, acc0[t], false, false);
        acc1[t] = __builtin_amdgcn_wmma_f32_16x16x32_bf16(
            false, a1, false, bB[t], (short)0, acc1[t], false, false);
      }
    }
  }

  // ---- fused epilogue:  relu(e + b_enc + d[b]) . W_full  ----
  const int b0 = row0 / L_;
  const int b1 = (row0 + ROWS_PER_WG - 1) / L_;   // at most 2 batches per 32 rows
  const int boundary = (b0 + 1) * L_ - row0;      // localRow >= boundary -> batch b1

  float be[4], wf[4], db0[4], db1[4];
#pragma unroll
  for (int t = 0; t < 4; ++t) {
    const int nt = nbase + t * 16 + mrow;
    be[t]  = b_enc[nt];
    wf[t]  = W_full[nt];
    db0[t] = dmat[b0 * ATT_ + nt];
    db1[t] = (b1 != b0) ? dmat[b1 * ATT_ + nt] : db0[t];
  }

  const int mhi = (lane >> 4) * 8;      // C layout: lanes>=16 hold rows M=8..15
  float rs0[8] = {}, rs1[8] = {};
#pragma unroll
  for (int t = 0; t < 4; ++t) {
#pragma unroll
    for (int j = 0; j < 8; ++j) {
      const int lr0 = mhi + j;
      const int lr1 = 16 + mhi + j;
      float v0 = acc0[t][j] + be[t] + ((lr0 >= boundary) ? db1[t] : db0[t]);
      float v1 = acc1[t][j] + be[t] + ((lr1 >= boundary) ? db1[t] : db0[t]);
      v0 = fmaxf(v0, 0.0f);
      v1 = fmaxf(v1, 0.0f);
      rs0[j] = fmaf(v0, wf[t], rs0[j]);
      rs1[j] = fmaf(v1, wf[t], rs1[j]);
    }
  }
  // reduce across the 16 lanes holding distinct N (xor masks stay in-half)
#pragma unroll
  for (int j = 0; j < 8; ++j) {
    float v0 = rs0[j], v1 = rs1[j];
#pragma unroll
    for (int m = 8; m >= 1; m >>= 1) {
      v0 += __shfl_xor(v0, m, 32);
      v1 += __shfl_xor(v1, m, 32);
    }
    if ((lane & 15) == 0) {             // lane 0 -> rows 0..7, lane 16 -> rows 8..15
      atomicAdd(&sAtt[mhi + j], v0);
      atomicAdd(&sAtt[16 + mhi + j], v1);
    }
  }
  __syncthreads();
  if (tid < ROWS_PER_WG)
    attRaw[row0 + tid] = sAtt[tid] + b_full[0];
}

// ---------------------------------------------------------------------------
// Kernel 4: per-batch softmax over L, alpha out, ctx = sum_l alpha*enc
// ---------------------------------------------------------------------------
__global__ __launch_bounds__(256)
void k_softmax_ctx(const float* __restrict__ enc,
                   const float* __restrict__ attRaw,
                   float* __restrict__ ctx,
                   float* __restrict__ alphaOut)
{
  __shared__ float sa[L_];
  __shared__ float red[256];
  const int b = blockIdx.x, t = threadIdx.x;

  const float v = (t < L_) ? attRaw[b * L_ + t] : -3.4e38f;
  red[t] = v;
  __syncthreads();
  for (int s = 128; s > 0; s >>= 1) {
    if (t < s) red[t] = fmaxf(red[t], red[t + s]);
    __syncthreads();
  }
  const float mx = red[0];
  __syncthreads();
  const float e = (t < L_) ? __expf(v - mx) : 0.0f;
  red[t] = e;
  __syncthreads();
  for (int s = 128; s > 0; s >>= 1) {
    if (t < s) red[t] += red[t + s];
    __syncthreads();
  }
  const float inv = 1.0f / red[0];
  const float al = e * inv;
  if (t < L_) { sa[t] = al; alphaOut[b * L_ + t] = al; }
  __syncthreads();

  float acc[8] = {};
  const float* encB = enc + (size_t)b * L_ * ENC_;
  for (int l = 0; l < L_; ++l) {
    const float a = sa[l];
    const float4* row = reinterpret_cast<const float4*>(encB + (size_t)l * ENC_);
    const float4 x0 = row[t * 2 + 0];
    const float4 x1 = row[t * 2 + 1];
    acc[0] = fmaf(a, x0.x, acc[0]); acc[1] = fmaf(a, x0.y, acc[1]);
    acc[2] = fmaf(a, x0.z, acc[2]); acc[3] = fmaf(a, x0.w, acc[3]);
    acc[4] = fmaf(a, x1.x, acc[4]); acc[5] = fmaf(a, x1.y, acc[5]);
    acc[6] = fmaf(a, x1.z, acc[6]); acc[7] = fmaf(a, x1.w, acc[7]);
  }
  float4* o = reinterpret_cast<float4*>(ctx + (size_t)b * ENC_);
  o[t * 2 + 0] = make_float4(acc[0], acc[1], acc[2], acc[3]);
  o[t * 2 + 1] = make_float4(acc[4], acc[5], acc[6], acc[7]);
}

// ---------------------------------------------------------------------------
extern "C" void kernel_launch(void* const* d_in, const int* in_sizes, int n_in,
                              void* d_out, int out_size, void* d_ws, size_t ws_size,
                              hipStream_t stream) {
  const float* enc    = (const float*)d_in[0];
  const float* dec    = (const float*)d_in[1];
  const float* W_enc  = (const float*)d_in[2];
  const float* b_enc  = (const float*)d_in[3];
  const float* W_dec  = (const float*)d_in[4];
  const float* b_dec  = (const float*)d_in[5];
  const float* W_full = (const float*)d_in[6];
  const float* b_full = (const float*)d_in[7];

  // workspace layout: wbT (2 MB) | dmat (512 KB) | attRaw (200 KB)
  unsigned short* wbT = (unsigned short*)d_ws;
  float* dmat   = (float*)((char*)d_ws + (size_t)ATT_ * ENC_ * 2);
  float* attRaw = (float*)((char*)d_ws + (size_t)ATT_ * ENC_ * 2 + (size_t)B_ * ATT_ * 4);

  // output: ctx (B*ENC floats) then alpha (B*L floats), concatenated
  float* ctx   = (float*)d_out;
  float* alpha = ctx + (size_t)B_ * ENC_;

  k_wenc_transpose<<<dim3(ATT_ / 32, ENC_ / 32), 256, 0, stream>>>(W_enc, wbT);
  k_dec_proj<<<B_, 256, 0, stream>>>(dec, W_dec, b_dec, dmat);
  k_enc_att<<<NTILES, 256, (size_t)ROWS_PER_WG * ASTRIDE * 2, stream>>>(
      enc, wbT, b_enc, dmat, W_full, b_full, attRaw);
  k_softmax_ctx<<<B_, 256, 0, stream>>>(enc, attRaw, ctx, alpha);
}
